// PainnModel_36971078484138
// MI455X (gfx1250) — matrix operand order repaired and checked by reading, not compiled, not run on hardware
//
#include <hip/hip_runtime.h>

// PaiNN forward for MI455X (gfx1250, wave32, WMMA).
// Input flattening assumption (setup_inputs dict order, params dict flattened in
// insertion order):
//  0 embedding(119x128) 1 msg_W1(3x128x128) 2 msg_b1(3x128) 3 msg_W2(3x128x384)
//  4 msg_b2(3x384) 5 flt_W(3x20x384) 6 flt_b(3x384) 7 upd_WU(3x128x128)
//  8 upd_bU(3x128) 9 upd_WV(3x128x128) 10 upd_bV(3x128) 11 upd_W1(3x256x128)
// 12 upd_b1(3x128) 13 upd_W2(3x128x384) 14 upd_b2(3x384) 15 ro_W1(128x128)
// 16 ro_b1(128) 17 ro_W2(128x1) 18 ro_b2(1) 19 z 20 edge 21 edge_diff
// 22 edge_dist 23 image_idx

#define F 128
#define EDGE_SZ 20
#define CUTOFF 5.0f
#define PI_F 3.14159265358979323846f
#define N_LAYERS 3

typedef __attribute__((ext_vector_type(16))) _Float16 v16h;
typedef __attribute__((ext_vector_type(8)))  float    v8f;

static __host__ int cdiv(int a, int b) { return (a + b - 1) / b; }

// Fast SiLU: x * rcp(1+exp(-x)) -> v_exp_f32 + v_rcp_f32 (no IEEE div sequence)
__device__ __forceinline__ float silu_f(float x) {
  return x * __builtin_amdgcn_rcpf(1.f + __expf(-x));
}

// ---------------------------------------------------------------------------
// Embedding gather + zero node vectors
// ---------------------------------------------------------------------------
__global__ void k_embed_init(const float* __restrict__ emb, const int* __restrict__ z,
                             float* __restrict__ ns, float* __restrict__ nv, int nNodes)
{
  int idx = blockIdx.x * blockDim.x + threadIdx.x;
  if (idx < nNodes * F) {
    int i = idx >> 7, f = idx & (F - 1);
    ns[idx] = emb[z[i] * F + f];
    nv[i * 3 * F + 0 * F + f] = 0.f;
    nv[i * 3 * F + 1 * F + f] = 0.f;
    nv[i * 3 * F + 2 * F + f] = 0.f;
  }
}

// ---------------------------------------------------------------------------
// WMMA GEMM: C[MxN] = act(A[MxK] @ W[KxN] + bias[N])
// block = 256 threads = 8 waves (2 x 4); block tile 32(M) x 128(N); K step 32.
// f32 -> f16 staged in LDS, f32 accumulate via v_wmma_f32_16x16x32_f16.
// Requires: K % 32 == 0, N % 128 == 0 (true for all uses: K in {128,256},
// N in {128,256,384}); M guarded.
// ---------------------------------------------------------------------------
#define BM 32
#define BN 128
#define BK 32
#define LDP (BK + 8)   // LDS row pad (halfs)

template <int ACT>
__global__ __launch_bounds__(256) void k_gemm_wmma(
    const float* __restrict__ A, const float* __restrict__ W,
    const float* __restrict__ bias, float* __restrict__ C,
    int M, int K, int N)
{
  __shared__ _Float16 As[BM][LDP];
  __shared__ _Float16 Bs[BN][LDP];

  const int tid  = threadIdx.x;
  const int lane = tid & 31;
  const int wave = tid >> 5;
  const int wm   = wave >> 2;   // 0..1
  const int wn   = wave & 3;    // 0..3
  const int row0 = blockIdx.x * BM;
  const int col0 = blockIdx.y * BN;

  v8f acc0 = {0.f,0.f,0.f,0.f,0.f,0.f,0.f,0.f};
  v8f acc1 = {0.f,0.f,0.f,0.f,0.f,0.f,0.f,0.f};

  for (int k0 = 0; k0 < K; k0 += BK) {
    // A tile: 32x32, coalesced over K
    for (int i = tid; i < BM * BK; i += 256) {
      int r = i >> 5, c = i & 31;
      int gr = row0 + r;
      As[r][c] = (_Float16)((gr < M) ? A[gr * K + k0 + c] : 0.f);
    }
    // B tile transposed into [n][k], coalesced over N
    for (int i = tid; i < BN * BK; i += 256) {
      int n = i & 127, c = i >> 7;
      Bs[n][c] = (_Float16)W[(k0 + c) * N + col0 + n];
    }
    __syncthreads();

    // A fragment (ISA 7.12.2, 16-bit A 16x32): lane m = lane&15, hs = lane>>4
    const int m  = wm * 16 + (lane & 15);
    const int hs = lane >> 4;
    v16h a;
#pragma unroll
    for (int v = 0; v < 16; ++v) a[v] = As[m][hs * 16 + v];

    // B fragments: column n, 16 contiguous K at hs*16
    const int nc = wn * 32 + (lane & 15);
    v16h b0, b1;
#pragma unroll
    for (int v = 0; v < 16; ++v) b0[v] = Bs[nc][hs * 16 + v];
#pragma unroll
    for (int v = 0; v < 16; ++v) b1[v] = Bs[nc + 16][hs * 16 + v];

    acc0 = __builtin_amdgcn_wmma_f32_16x16x32_f16(false, a, false, b0,
                                                  (short)0, acc0, false, false);
    acc1 = __builtin_amdgcn_wmma_f32_16x16x32_f16(false, a, false, b1,
                                                  (short)0, acc1, false, false);
    __syncthreads();
  }

  // C/D layout: vgpr r -> m = r + 8*(lane>>4); n = lane&15
  const int nlo  = lane & 15;
  const int rsel = lane >> 4;
#pragma unroll
  for (int r = 0; r < 8; ++r) {
    int mrow = row0 + wm * 16 + r + 8 * rsel;
    if (mrow < M) {
      int c0 = col0 + wn * 32 + nlo;
      float v0 = acc0[r] + bias[c0];
      float v1 = acc1[r] + bias[c0 + 16];
      if (ACT) { v0 = silu_f(v0); v1 = silu_f(v1); }
      C[mrow * N + c0]      = v0;
      C[mrow * N + c0 + 16] = v1;
    }
  }
}

// ---------------------------------------------------------------------------
// Fused edge-message kernel. Filter weights (20x384 = 30KB) LDS-resident.
// 256 threads = 2 groups x 128; each group processes EPG edges.
// Per edge: rbf/fcut/unit in-kernel (one v_rcp_f32 for 1/d), fw = rbf@fltW+b,
// fo = fw*fcut*so[send], scatter-add msg_s -> ns[recv], msg_v -> nv[recv].
// ---------------------------------------------------------------------------
#define EPG 16
__global__ __launch_bounds__(256) void k_edge_message(
    const int* __restrict__ edge, const float* __restrict__ edge_diff,
    const float* __restrict__ edge_dist, const float* __restrict__ so,
    const float* __restrict__ fltW, const float* __restrict__ fltb,
    float* __restrict__ ns, float* __restrict__ nv, int nEdges)
{
  __shared__ float sW[EDGE_SZ][3 * F];
  __shared__ float sb[3 * F];
  __shared__ float srbf[2][EDGE_SZ];
  __shared__ float sunit[2][3];
  __shared__ float sfc[2];
  __shared__ int   ssr[2][2];   // [grp][0]=recv, [1]=send

  const int tid = threadIdx.x;
  const int grp = tid >> 7;
  const int j   = tid & 127;

  for (int i = tid; i < EDGE_SZ * 3 * F; i += 256)
    sW[i / (3 * F)][i % (3 * F)] = fltW[i];
  for (int i = tid; i < 3 * F; i += 256) sb[i] = fltb[i];
  __syncthreads();

  const int base = blockIdx.x * (2 * EPG);
  for (int it = 0; it < EPG; ++it) {
    const int e = base + it * 2 + grp;
    if (e < nEdges) {
      float d  = edge_dist[e];
      float rd = __builtin_amdgcn_rcpf(d);
      if (j < EDGE_SZ) {
        srbf[grp][j] = __sinf(d * (float)(j + 1) * (PI_F / CUTOFF)) * rd;
      } else if (j < EDGE_SZ + 3) {
        sunit[grp][j - EDGE_SZ] = edge_diff[e * 3 + (j - EDGE_SZ)] * rd;
      } else if (j == 24) {
        sfc[grp] = (d < CUTOFF) ? 0.5f * (__cosf(PI_F * d / CUTOFF) + 1.f) : 0.f;
      } else if (j == 25) {
        ssr[grp][0] = edge[e * 2 + 0];
      } else if (j == 26) {
        ssr[grp][1] = edge[e * 2 + 1];
      }
    }
    __syncthreads();
    if (e < nEdges) {
      float w0 = sb[j], w1 = sb[j + F], w2 = sb[j + 2 * F];
#pragma unroll
      for (int k = 0; k < EDGE_SZ; ++k) {
        float r = srbf[grp][k];
        w0 += r * sW[k][j];
        w1 += r * sW[k][j + F];
        w2 += r * sW[k][j + 2 * F];
      }
      const float fc  = sfc[grp];
      const int   rcv = ssr[grp][0];
      const int   snd = ssr[grp][1];
      const float gsv = w0 * fc * so[snd * (3 * F) + j];
      const float gev = w1 * fc * so[snd * (3 * F) + F + j];
      const float msg = w2 * fc * so[snd * (3 * F) + 2 * F + j];
      atomicAdd(&ns[rcv * F + j], msg);
#pragma unroll
      for (int dd = 0; dd < 3; ++dd) {
        float mv = nv[snd * 3 * F + dd * F + j] * gsv + gev * sunit[grp][dd];
        atomicAdd(&nv[rcv * 3 * F + dd * F + j], mv);
      }
    }
    __syncthreads();
  }
}

// ---------------------------------------------------------------------------
// mlp_in = [ ||Vv||_dim , ns ]  (nNodes x 256)
// ---------------------------------------------------------------------------
__global__ void k_vn_concat(const float* __restrict__ Vv, const float* __restrict__ ns,
                            float* __restrict__ mlp_in, int nNodes)
{
  int idx = blockIdx.x * blockDim.x + threadIdx.x;
  if (idx < nNodes * F) {
    int i = idx >> 7, n = idx & (F - 1);
    float v0 = Vv[i * 3 * F + n];
    float v1 = Vv[i * 3 * F + F + n];
    float v2 = Vv[i * 3 * F + 2 * F + n];
    mlp_in[i * 2 * F + n]     = sqrtf(v0 * v0 + v1 * v1 + v2 * v2);
    mlp_in[i * 2 * F + F + n] = ns[idx];
  }
}

// ---------------------------------------------------------------------------
// Node update: nv += a_vv*Uv ; ns += a_sv*dot(Uv,Vv) + a_ss
// ---------------------------------------------------------------------------
__global__ void k_node_update(const float* __restrict__ Uv, const float* __restrict__ Vv,
                              const float* __restrict__ mo, float* __restrict__ ns,
                              float* __restrict__ nv, int nNodes)
{
  int idx = blockIdx.x * blockDim.x + threadIdx.x;
  if (idx < nNodes * F) {
    int i = idx >> 7, n = idx & (F - 1);
    float u0 = Uv[i * 3 * F + n],          v0 = Vv[i * 3 * F + n];
    float u1 = Uv[i * 3 * F + F + n],      v1 = Vv[i * 3 * F + F + n];
    float u2 = Uv[i * 3 * F + 2 * F + n],  v2 = Vv[i * 3 * F + 2 * F + n];
    float avv = mo[i * 3 * F + n];
    float asv = mo[i * 3 * F + F + n];
    float ass = mo[i * 3 * F + 2 * F + n];
    float dot = u0 * v0 + u1 * v1 + u2 * v2;
    ns[idx] += asv * dot + ass;
    nv[i * 3 * F + n]         += avv * u0;
    nv[i * 3 * F + F + n]     += avv * u1;
    nv[i * 3 * F + 2 * F + n] += avv * u2;
  }
}

// ---------------------------------------------------------------------------
// Readout: atom_e = h@ro_W2 + ro_b2; segment-sum into out[image_idx]
// ---------------------------------------------------------------------------
__global__ void k_zero(float* __restrict__ p, int n)
{
  int i = blockIdx.x * blockDim.x + threadIdx.x;
  if (i < n) p[i] = 0.f;
}

__global__ void k_readout(const float* __restrict__ h, const float* __restrict__ roW2,
                          const float* __restrict__ rob2, const int* __restrict__ image_idx,
                          float* __restrict__ out, int nNodes)
{
  int i = blockIdx.x * blockDim.x + threadIdx.x;
  if (i < nNodes) {
    float acc = rob2[0];
#pragma unroll 8
    for (int k = 0; k < F; ++k) acc += h[i * F + k] * roW2[k];
    atomicAdd(&out[image_idx[i]], acc);
  }
}

// ---------------------------------------------------------------------------
extern "C" void kernel_launch(void* const* d_in, const int* in_sizes, int n_in,
                              void* d_out, int out_size, void* d_ws, size_t ws_size,
                              hipStream_t stream)
{
  (void)n_in; (void)ws_size;
  const float* emb    = (const float*)d_in[0];
  const float* msgW1  = (const float*)d_in[1];
  const float* msgB1  = (const float*)d_in[2];
  const float* msgW2  = (const float*)d_in[3];
  const float* msgB2  = (const float*)d_in[4];
  const float* fltW   = (const float*)d_in[5];
  const float* fltB   = (const float*)d_in[6];
  const float* updWU  = (const float*)d_in[7];
  const float* updBU  = (const float*)d_in[8];
  const float* updWV  = (const float*)d_in[9];
  const float* updBV  = (const float*)d_in[10];
  const float* updW1  = (const float*)d_in[11];
  const float* updB1  = (const float*)d_in[12];
  const float* updW2  = (const float*)d_in[13];
  const float* updB2  = (const float*)d_in[14];
  const float* roW1   = (const float*)d_in[15];
  const float* roB1   = (const float*)d_in[16];
  const float* roW2   = (const float*)d_in[17];
  const float* roB2   = (const float*)d_in[18];
  const int*   z      = (const int*)d_in[19];
  const int*   edge   = (const int*)d_in[20];
  const float* ediff  = (const float*)d_in[21];
  const float* edist  = (const float*)d_in[22];
  const int*   imgIdx = (const int*)d_in[23];

  const int nNodes = in_sizes[19];          // 12500
  const int nEdges = in_sizes[22];          // 200000
  float* out = (float*)d_out;

  // Workspace layout (floats). Node state stays L2-resident (192 MB).
  float* ws      = (float*)d_ws;
  float* ns      = ws;                               // nNodes*F
  float* nv      = ns + (size_t)nNodes * F;          // nNodes*3F
  float* hbuf    = nv + (size_t)nNodes * 3 * F;      // nNodes*F   (h1/h2/h3)
  float* soUv    = hbuf + (size_t)nNodes * F;        // nNodes*3F  (so, then Uv)
  float* Vv      = soUv + (size_t)nNodes * 3 * F;    // nNodes*3F
  float* mlp_in  = Vv + (size_t)nNodes * 3 * F;      // nNodes*2F
  float* mlp_out = mlp_in + (size_t)nNodes * 2 * F;  // nNodes*3F

  const int elemBlocks = cdiv(nNodes * F, 256);
  const dim3 blk(256);

  k_embed_init<<<elemBlocks, blk, 0, stream>>>(emb, z, ns, nv, nNodes);

  for (int l = 0; l < N_LAYERS; ++l) {
    const float* mW1 = msgW1 + (size_t)l * F * F;
    const float* mB1 = msgB1 + (size_t)l * F;
    const float* mW2 = msgW2 + (size_t)l * F * 3 * F;
    const float* mB2 = msgB2 + (size_t)l * 3 * F;
    const float* fW  = fltW  + (size_t)l * EDGE_SZ * 3 * F;
    const float* fB  = fltB  + (size_t)l * 3 * F;
    const float* uWU = updWU + (size_t)l * F * F;
    const float* uBU = updBU + (size_t)l * F;
    const float* uWV = updWV + (size_t)l * F * F;
    const float* uBV = updBV + (size_t)l * F;
    const float* uW1 = updW1 + (size_t)l * 2 * F * F;
    const float* uB1 = updB1 + (size_t)l * F;
    const float* uW2 = updW2 + (size_t)l * F * 3 * F;
    const float* uB2 = updB2 + (size_t)l * 3 * F;

    // h1 = silu(ns @ msg_W1 + b1)        [nNodes x 128]
    k_gemm_wmma<1><<<dim3(cdiv(nNodes, BM), F / BN), blk, 0, stream>>>(
        ns, mW1, mB1, hbuf, nNodes, F, F);
    // so = h1 @ msg_W2 + b2              [nNodes x 384]
    k_gemm_wmma<0><<<dim3(cdiv(nNodes, BM), 3 * F / BN), blk, 0, stream>>>(
        hbuf, mW2, mB2, soUv, nNodes, F, 3 * F);
    // fused edge messages -> ns, nv (atomics)
    k_edge_message<<<cdiv(nEdges, 2 * EPG), blk, 0, stream>>>(
        edge, ediff, edist, soUv, fW, fB, ns, nv, nEdges);
    // Uv = nv @ upd_WU + bU              [(3*nNodes) x 128]
    k_gemm_wmma<0><<<dim3(cdiv(3 * nNodes, BM), F / BN), blk, 0, stream>>>(
        nv, uWU, uBU, soUv, 3 * nNodes, F, F);
    // Vv = nv @ upd_WV + bV
    k_gemm_wmma<0><<<dim3(cdiv(3 * nNodes, BM), F / BN), blk, 0, stream>>>(
        nv, uWV, uBV, Vv, 3 * nNodes, F, F);
    // mlp_in = [||Vv||, ns]
    k_vn_concat<<<elemBlocks, blk, 0, stream>>>(Vv, ns, mlp_in, nNodes);
    // h2 = silu(mlp_in @ upd_W1 + b1)    [nNodes x 128], K=256
    k_gemm_wmma<1><<<dim3(cdiv(nNodes, BM), F / BN), blk, 0, stream>>>(
        mlp_in, uW1, uB1, hbuf, nNodes, 2 * F, F);
    // mlp_out = h2 @ upd_W2 + b2         [nNodes x 384]
    k_gemm_wmma<0><<<dim3(cdiv(nNodes, BM), 3 * F / BN), blk, 0, stream>>>(
        hbuf, uW2, uB2, mlp_out, nNodes, F, 3 * F);
    // gated residual updates
    k_node_update<<<elemBlocks, blk, 0, stream>>>(soUv, Vv, mlp_out, ns, nv, nNodes);
  }

  // readout: h3 = silu(ns @ ro_W1 + b1)
  k_gemm_wmma<1><<<dim3(cdiv(nNodes, BM), F / BN), blk, 0, stream>>>(
      ns, roW1, roB1, hbuf, nNodes, F, F);
  k_zero<<<1, 32, 0, stream>>>(out, out_size);
  k_readout<<<cdiv(nNodes, 256), blk, 0, stream>>>(hbuf, roW2, roB2, imgIdx, out, nNodes);
}
